// Head_50646254354523
// MI455X (gfx1250) — matrix operand order repaired and checked by reading, not compiled, not run on hardware
//
#include <hip/hip_runtime.h>

// ---------------------------------------------------------------------------
// Causal attention head, column-softmax (softmax over query axis).
// B=8 T=2048 E=1024 HS=128.  Strategy:
//   K0: Wq/Wk/Wv fp32 -> bf16 (once; removes conversions from GEMM hot loop)
//   K1: q,k = x*Wq^T, x*Wk^T (bf16, row-major), vT = (x*Wv^T)^T (bf16)
//   K2: colsum[b,k] = sum_{q>=k} exp(S[q,k]/32)   (S recomputed via WMMA)
//   K3: out[q,h] = sum_k exp(S[q,k]/32)/colsum[k] * v[k,h]  (S recomputed)
// All matmuls use v_wmma_f32_16x16x32_bf16 (f32 accumulate).
// Workspace: q 4MB + k 4MB + vT 4MB + colsum 64KB + Wbf 768KB (~13.4MB, L2-resident).
// ---------------------------------------------------------------------------

#define BATCH 8
#define TSEQ  2048
#define EMB   1024
#define HSZ   128
#define SCALE 0.03125f   // E^-0.5 = 1/32

typedef __attribute__((ext_vector_type(16))) __bf16 v16bf;
typedef __attribute__((ext_vector_type(8)))  float  v8f;
typedef __attribute__((ext_vector_type(4)))  float  f32x4;
typedef __attribute__((ext_vector_type(4)))  unsigned int u32x4;

union F16x { u32x4 q[2]; v16bf v; };

__device__ __forceinline__ v8f wmma_acc(v16bf a, v16bf b, v8f c) {
  // D = A(16x32 bf16) * B(32x16 bf16) + C(16x16 f32)
  return __builtin_amdgcn_wmma_f32_16x16x32_bf16(
      /*neg_a=*/false, a, /*neg_b=*/false, b,
      /*c_mod=*/(short)0, c, /*reuse_a=*/false, /*reuse_b=*/false);
}

// ---------------- Kernel 0: weight conversion fp32 -> bf16 -----------------
__global__ __launch_bounds__(256) void wcvt_kernel(
    const float* __restrict__ Wq, const float* __restrict__ Wk,
    const float* __restrict__ Wv, __bf16* __restrict__ Wbf)  // [3][HSZ][EMB]
{
  const int MAT4 = (HSZ * EMB) / 4;                  // float4s per matrix
  const int i = blockIdx.x * 256 + threadIdx.x;      // exactly 3*MAT4 threads
  const int mat = i / MAT4;
  const int off = (i - mat * MAT4) * 4;
  const float* src = (mat == 0) ? Wq : (mat == 1 ? Wk : Wv);
  f32x4 f = *(const f32x4*)(src + off);
  __bf16* d = Wbf + (size_t)mat * HSZ * EMB + off;
  d[0] = (__bf16)f.x; d[1] = (__bf16)f.y; d[2] = (__bf16)f.z; d[3] = (__bf16)f.w;
}

// ---------------- Kernel 1: QKV projection (fp32 x, bf16 W -> bf16 out) ----
__global__ __launch_bounds__(128) void qkv_kernel(
    const float* __restrict__ x, const __bf16* __restrict__ Wbf,
    __bf16* __restrict__ qbf, __bf16* __restrict__ kbf,
    __bf16* __restrict__ vT)
{
  const int lane = threadIdx.x & 31;
  const int wid  = threadIdx.x >> 5;
  const int task = blockIdx.x * 4 + wid;            // 3 * 1024 tasks
  const int MT   = (BATCH * TSEQ) / 16;             // 1024 row tiles
  const int which = task / MT;                      // 0:q 1:k 2:v
  const int m0    = (task % MT) * 16;
  const int hlf = lane >> 4, l15 = lane & 15;
  const int abase = 8 * hlf;
  const __bf16* W = Wbf + (size_t)which * HSZ * EMB;

  v8f acc[8] = {};
  const float* xrow = x + (size_t)(m0 + l15) * EMB;

  for (int k0 = 0; k0 < EMB; k0 += 32) {
    // A fragment: row m0+l15, K pattern {16*hlf..16*hlf+15}
    v16bf af;
    {
      const f32x4* p0 = (const f32x4*)(xrow + k0 + 2 * abase);
      const f32x4* p1 = (const f32x4*)(xrow + k0 + 2 * abase + 8);
      f32x4 f;
      f = p0[0]; af[0]=(__bf16)f.x; af[1]=(__bf16)f.y; af[2]=(__bf16)f.z; af[3]=(__bf16)f.w;
      f = p0[1]; af[4]=(__bf16)f.x; af[5]=(__bf16)f.y; af[6]=(__bf16)f.z; af[7]=(__bf16)f.w;
      f = p1[0]; af[8]=(__bf16)f.x; af[9]=(__bf16)f.y; af[10]=(__bf16)f.z; af[11]=(__bf16)f.w;
      f = p1[1]; af[12]=(__bf16)f.x; af[13]=(__bf16)f.y; af[14]=(__bf16)f.z; af[15]=(__bf16)f.w;
    }
    if (k0 + 256 < EMB) __builtin_prefetch(xrow + k0 + 256, 0, 0);
#pragma unroll
    for (int nt = 0; nt < 8; ++nt) {
      const int n = nt * 16 + l15;                  // B column = W row n
      const u32x4* wp = (const u32x4*)(W + (size_t)n * EMB + k0 + 16 * hlf);
      F16x bfrag; bfrag.q[0] = wp[0]; bfrag.q[1] = wp[1];
      acc[nt] = wmma_acc(af, bfrag.v, acc[nt]);
    }
  }

  if (which < 2) {
    __bf16* dst = (which == 0) ? qbf : kbf;
#pragma unroll
    for (int nt = 0; nt < 8; ++nt) {
      const int n = nt * 16 + l15;
#pragma unroll
      for (int v = 0; v < 8; ++v)
        dst[(size_t)(m0 + v + 8 * hlf) * HSZ + n] = (__bf16)acc[nt][v];
    }
  } else {                                          // v stored transposed
#pragma unroll
    for (int nt = 0; nt < 8; ++nt) {
      const int h = nt * 16 + l15;
#pragma unroll
      for (int v = 0; v < 8; ++v) {
        const int tg = m0 + v + 8 * hlf;
        const int b = tg >> 11, t = tg & (TSEQ - 1);
        vT[((size_t)b * HSZ + h) * TSEQ + t] = (__bf16)acc[nt][v];
      }
    }
  }
}

// ---------------- Kernel 2: column sums of exp(masked scores) --------------
__global__ __launch_bounds__(128) void colsum_kernel(
    const __bf16* __restrict__ qbf, const __bf16* __restrict__ kbf,
    float* __restrict__ colsum)
{
  const int lane = threadIdx.x & 31;
  const int wid  = threadIdx.x >> 5;
  const int task = blockIdx.x * 4 + wid;            // 8 * 128 tasks
  const int b  = task >> 7;
  const int kt = task & 127;
  const int k0 = kt * 16;
  const int hlf = lane >> 4, l15 = lane & 15;
  const int abase = 8 * hlf;
  const int col = k0 + l15;

  // B fragments: k row (= S column) contiguous along h; preload 4 h-chunks
  const __bf16* kb = kbf + ((size_t)b * TSEQ + col) * HSZ;
  F16x bf[4];
#pragma unroll
  for (int hc = 0; hc < 4; ++hc) {
    const u32x4* p = (const u32x4*)(kb + hc * 32 + 16 * hlf);
    bf[hc].q[0] = p[0]; bf[hc].q[1] = p[1];
  }

  float sum = 0.f;
  for (int qt = kt; qt < TSEQ / 16; ++qt) {
    const int q0 = qt * 16;
    const __bf16* qrow = qbf + ((size_t)b * TSEQ + q0 + l15) * HSZ;
    v8f s = {};
#pragma unroll
    for (int hc = 0; hc < 4; ++hc) {
      F16x af;
      af.q[0] = *(const u32x4*)(qrow + hc * 32 + 2 * abase);
      af.q[1] = *(const u32x4*)(qrow + hc * 32 + 2 * abase + 8);
      s = wmma_acc(af.v, bf[hc].v, s);
    }
#pragma unroll
    for (int v = 0; v < 8; ++v) {
      const int row = q0 + v + 8 * hlf;
      if (row >= col) sum += __expf(s[v] * SCALE);
    }
  }
  sum += __shfl_xor(sum, 16);                        // fold half-waves
  if (hlf == 0) colsum[(size_t)b * TSEQ + col] = sum;
}

// ---------------- Kernel 3: out = softmax_col(S) @ V -----------------------
__global__ __launch_bounds__(128) void attn_out_kernel(
    const __bf16* __restrict__ qbf, const __bf16* __restrict__ kbf,
    const __bf16* __restrict__ vT, const float* __restrict__ colsum,
    float* __restrict__ out)
{
  __shared__ __bf16 ldsP[4][16 * 32];                // per-wave P tile
  const int lane = threadIdx.x & 31;
  const int wid  = threadIdx.x >> 5;
  const int task = blockIdx.x * 4 + wid;             // 8 * 128 tasks
  const int b  = task >> 7;
  const int qt = task & 127;
  const int q0 = qt * 16;
  const int hlf = lane >> 4, l15 = lane & 15;
  const int abase = 8 * hlf;
  __bf16* lds = ldsP[wid];

  // preload Q A-fragments (4 h-chunks)
  const __bf16* qrow = qbf + ((size_t)b * TSEQ + q0 + l15) * HSZ;
  F16x aq[4];
#pragma unroll
  for (int hc = 0; hc < 4; ++hc) {
    aq[hc].q[0] = *(const u32x4*)(qrow + hc * 32 + 2 * abase);
    aq[hc].q[1] = *(const u32x4*)(qrow + hc * 32 + 2 * abase + 8);
  }

  v8f accO[8] = {};
  const int nch = (q0 + 16 + 31) >> 5;               // 32-wide k chunks

  for (int c = 0; c < nch; ++c) {
    const int k0 = c * 32;
    // ---- two 16x16 score tiles -> probabilities -> LDS (bf16, 16x32)
#pragma unroll
    for (int sub = 0; sub < 2; ++sub) {
      const int col = k0 + sub * 16 + l15;
      const __bf16* kb = kbf + ((size_t)b * TSEQ + col) * HSZ;
      v8f s = {};
#pragma unroll
      for (int hc = 0; hc < 4; ++hc) {
        F16x bfrag;
        const u32x4* p = (const u32x4*)(kb + hc * 32 + 16 * hlf);
        bfrag.q[0] = p[0]; bfrag.q[1] = p[1];
        s = wmma_acc(aq[hc].v, bfrag.v, s);
      }
      const float rs = 1.0f / colsum[(size_t)b * TSEQ + col];
#pragma unroll
      for (int v = 0; v < 8; ++v) {
        const int row = q0 + v + 8 * hlf;
        const float pv = (row >= col) ? __expf(s[v] * SCALE) * rs : 0.f;
        lds[(v + 8 * hlf) * 32 + sub * 16 + l15] = (__bf16)pv;
      }
    }
    // ---- LDS transpose-read: P as A fragment (row = l15)
    v16bf pa;
    {
      const __bf16* lp = lds + l15 * 32;
#pragma unroll
      for (int e = 0; e < 16; ++e) {
        const int kk = 2 * abase + e;
        pa[e] = lp[kk];
      }
    }
    // ---- out += P(16x32) @ V(32x128);  vT gives contiguous B columns
#pragma unroll
    for (int ht = 0; ht < 8; ++ht) {
      const int h = ht * 16 + l15;
      const u32x4* p = (const u32x4*)(vT + ((size_t)b * HSZ + h) * TSEQ + k0 + 16 * hlf);
      F16x bv; bv.q[0] = p[0]; bv.q[1] = p[1];
      accO[ht] = wmma_acc(pa, bv.v, accO[ht]);
    }
  }

#pragma unroll
  for (int ht = 0; ht < 8; ++ht) {
    const int h = ht * 16 + l15;
#pragma unroll
    for (int v = 0; v < 8; ++v) {
      const int row = q0 + v + 8 * hlf;
      out[((size_t)b * TSEQ + row) * HSZ + h] = accO[ht][v];
    }
  }
}

// ---------------------------------------------------------------------------
extern "C" void kernel_launch(void* const* d_in, const int* in_sizes, int n_in,
                              void* d_out, int out_size, void* d_ws, size_t ws_size,
                              hipStream_t stream) {
  (void)in_sizes; (void)n_in; (void)out_size; (void)ws_size;
  const float* x  = (const float*)d_in[0];
  const float* Wk = (const float*)d_in[1];
  const float* Wq = (const float*)d_in[2];
  const float* Wv = (const float*)d_in[3];
  float* out = (float*)d_out;

  char* ws = (char*)d_ws;
  const size_t QKV_BYTES = (size_t)BATCH * TSEQ * HSZ * sizeof(__bf16); // 4MB
  __bf16* qbf = (__bf16*)(ws);
  __bf16* kbf = (__bf16*)(ws + QKV_BYTES);
  __bf16* vT  = (__bf16*)(ws + 2 * QKV_BYTES);
  float* colsum = (float*)(ws + 3 * QKV_BYTES);
  __bf16* Wbf = (__bf16*)(ws + 3 * QKV_BYTES + (size_t)BATCH * TSEQ * sizeof(float));

  const dim3 blk128(128);
  // 3*HSZ*EMB/4 float4s / 256 threads = 384 blocks (exact)
  wcvt_kernel<<<dim3((3 * HSZ * EMB / 4) / 256), dim3(256), 0, stream>>>(Wq, Wk, Wv, Wbf);
  qkv_kernel<<<dim3(768), blk128, 0, stream>>>(x, Wbf, qbf, kbf, vT);
  colsum_kernel<<<dim3(256), blk128, 0, stream>>>(qbf, kbf, colsum);
  attn_out_kernel<<<dim3(256), blk128, 0, stream>>>(qbf, kbf, vT, colsum, out);
}
